// Attention_4415226380830
// MI455X (gfx1250) — matrix operand order repaired and checked by reading, not compiled, not run on hardware
//
#include <hip/hip_runtime.h>
#include <hip/hip_fp16.h>

// ---------------------------------------------------------------------------
// Restormer-style channel attention for MI455X (gfx1250, wave32, WMMA + TDM).
// BW-bound workload (~90 GFLOP vs ~1.6 GB min traffic @ 23.3 TB/s -> ~70us):
//  - all intermediates f16, all contractions on v_wmma_f32_16x16x32_f16
//  - Gram (K=65536) streamed via Tensor Data Mover into double-buffered LDS
//  - 128-bit global loads on every streaming path
// ---------------------------------------------------------------------------

#define DIMC   192
#define HEADS  4
#define HEAD_DIM 48
#define BATCH  4
#define HGT    256
#define WID    256
#define HW     (HGT * WID)     // 65536
#define C3     (3 * DIMC)      // 576

typedef __attribute__((ext_vector_type(16))) _Float16 v16h;
typedef __attribute__((ext_vector_type(8)))  float    v8f;
typedef __attribute__((ext_vector_type(4)))  unsigned int u32x4;
typedef __attribute__((ext_vector_type(8)))  int      i32x8;
typedef __attribute__((ext_vector_type(4)))  int      i32x4;

#if defined(__has_builtin)
#if __has_builtin(__builtin_amdgcn_tensor_load_to_lds)
#define HAVE_TDM 1
#endif
#endif
#ifndef HAVE_TDM
#define HAVE_TDM 0
#endif

union FragU { uint4 q[2]; v16h v; };
union FragH { uint4 q; _Float16 h[8]; };

// A-fragment (16x32 f16, MxK). Lanes 0-15: K {0-7,16-23}; lanes 16-31: +8.
static __device__ __forceinline__ v16h load_a_frag(const _Float16* base,
                                                   int stride, int lane) {
  int m  = lane & 15;
  int kb = (lane < 16) ? 0 : 8;
  const _Float16* p = base + (size_t)m * stride + kb;
  FragU f;
  f.q[0] = *(const uint4*)(p);
  f.q[1] = *(const uint4*)(p + 16);
  return f.v;
}

// B-fragment (32x16 f16, KxN) from transposed tile Bt[n][k] (k contiguous).
static __device__ __forceinline__ v16h load_b_frag(const _Float16* baseT,
                                                   int stride, int lane) {
  int n  = lane & 15;
  int kb = (lane < 16) ? 0 : 8;
  const _Float16* p = baseT + (size_t)n * stride + kb;
  FragU f;
  f.q[0] = *(const uint4*)(p);
  f.q[1] = *(const uint4*)(p + 16);
  return f.v;
}

#define WMMA_F16(a, b, c) \
  __builtin_amdgcn_wmma_f32_16x16x32_f16(false, (a), false, (b), (short)0, (c), false, false)

// ---------------------------------------------------------------------------
// K0: zero the atomically-accumulated buffers (norms^2 and Gram S).
// ---------------------------------------------------------------------------
__global__ void zero_kernel(float* __restrict__ p, int n) {
  int i = blockIdx.x * 256 + threadIdx.x;
  if (i < n) p[i] = 0.f;
}

// ---------------------------------------------------------------------------
// K1: qkv 1x1 conv == GEMM  out[b, oc, p] = sum_c W[oc,c] * x[b,c,p]
// 32 oc x 64 px block tile, 8 waves x one 16x16 WMMA tile, K=192 (6x32).
// ---------------------------------------------------------------------------
__global__ __launch_bounds__(256) void qkv1x1_kernel(
    const float* __restrict__ x, const float* __restrict__ w,
    _Float16* __restrict__ out) {
  __shared__ alignas(16) _Float16 wl[32][DIMC + 8];
  __shared__ alignas(16) _Float16 xt[64][32 + 8];
  const int tid  = threadIdx.x;
  const int lane = tid & 31;
  const int wv   = tid >> 5;
  const int wr   = wv >> 2;
  const int wc   = wv & 3;
  const int p0   = blockIdx.x * 64;
  const int oc0  = blockIdx.y * 32;
  const int b    = blockIdx.z;

  for (int i = tid; i < 32 * (DIMC / 4); i += 256) {   // float4 weight loads
    int r = i / (DIMC / 4), c4 = (i % (DIMC / 4)) * 4;
    float4 v = *(const float4*)(w + (size_t)(oc0 + r) * DIMC + c4);
    wl[r][c4 + 0] = (_Float16)v.x;
    wl[r][c4 + 1] = (_Float16)v.y;
    wl[r][c4 + 2] = (_Float16)v.z;
    wl[r][c4 + 3] = (_Float16)v.w;
  }
  const float* xb = x + (size_t)b * DIMC * HW;

  v8f acc = {};
  for (int k0 = 0; k0 < DIMC; k0 += 32) {
    __syncthreads();
    {  // 128-bit coalesced reads, transposed LDS writes
      int f4 = tid & 15;          // 16 float4 per 64-px row
      int kk = tid >> 4;          // 0..15
      #pragma unroll
      for (int rep = 0; rep < 2; ++rep) {
        int k = kk + rep * 16;
        float4 v = *(const float4*)(xb + (size_t)(k0 + k) * HW + p0 + f4 * 4);
        xt[f4 * 4 + 0][k] = (_Float16)v.x;
        xt[f4 * 4 + 1][k] = (_Float16)v.y;
        xt[f4 * 4 + 2][k] = (_Float16)v.z;
        xt[f4 * 4 + 3][k] = (_Float16)v.w;
      }
    }
    __syncthreads();
    v16h a  = load_a_frag(&wl[wr * 16][k0], DIMC + 8, lane);
    v16h bb = load_b_frag(&xt[wc * 16][0], 32 + 8, lane);
    acc = WMMA_F16(a, bb, acc);
  }
  const int m_add = (lane < 16) ? 0 : 8;
  const int n     = lane & 15;
  size_t obase = ((size_t)b * C3 + oc0 + wr * 16) * HW + p0 + wc * 16 + n;
  #pragma unroll
  for (int r = 0; r < 8; ++r)
    out[obase + (size_t)(r + m_add) * HW] = (_Float16)acc[r];
}

// ---------------------------------------------------------------------------
// K2: depthwise 3x3 (SAME, zero pad), f16 in/out, f32 taps; fuses the
// spatial sum-of-squares for q/k L2 norms (atomicAdd f32).
// ---------------------------------------------------------------------------
__global__ __launch_bounds__(256) void dwconv_kernel(
    const _Float16* __restrict__ in, const float* __restrict__ dw,
    _Float16* __restrict__ out, float* __restrict__ norms2) {
  __shared__ _Float16 tile[34][36];
  __shared__ float red[8];
  const int ch = blockIdx.y, b = blockIdx.z;
  const int tx0 = (blockIdx.x % (WID / 32)) * 32;
  const int ty0 = (blockIdx.x / (WID / 32)) * 32;
  const _Float16* ip = in + ((size_t)b * C3 + ch) * HW;
  _Float16* op       = out + ((size_t)b * C3 + ch) * HW;
  float wgt[9];
  #pragma unroll
  for (int i = 0; i < 9; ++i) wgt[i] = dw[ch * 9 + i];

  for (int i = threadIdx.x; i < 34 * 34; i += 256) {
    int yy = i / 34, xx = i % 34;
    int gy = ty0 + yy - 1, gx = tx0 + xx - 1;
    _Float16 v = (_Float16)0.f;
    if (gy >= 0 && gy < HGT && gx >= 0 && gx < WID) v = ip[gy * WID + gx];
    tile[yy][xx] = v;
  }
  __syncthreads();
  float ss = 0.f;
  for (int i = threadIdx.x; i < 32 * 32; i += 256) {
    int yy = i >> 5, xx = i & 31;
    float acc = 0.f;
    #pragma unroll
    for (int dy = 0; dy < 3; ++dy)
      #pragma unroll
      for (int dx = 0; dx < 3; ++dx)
        acc += wgt[dy * 3 + dx] * (float)tile[yy + dy][xx + dx];
    op[(ty0 + yy) * WID + tx0 + xx] = (_Float16)acc;
    ss += acc * acc;
  }
  if (ch < 2 * DIMC) {
    #pragma unroll
    for (int off = 16; off > 0; off >>= 1) ss += __shfl_down(ss, off);
    if ((threadIdx.x & 31) == 0) red[threadIdx.x >> 5] = ss;
    __syncthreads();
    if (threadIdx.x == 0) {
      float t = 0.f;
      #pragma unroll
      for (int i = 0; i < 8; ++i) t += red[i];
      atomicAdd(&norms2[b * 2 * DIMC + ch], t);
    }
  }
}

// ---------------------------------------------------------------------------
// K3: Gram S[bh,d,e] = sum_n q[d,n]*k[e,n].  TDM streams 48x256 f16 tiles of
// q and k into double-buffered LDS (D# per ISA ch.8: data_size=2B, pad 16B
// per 512B row -> 264-half row stride, 16B-aligned rows). Wave w computes a
// 32-col slab of all nine 16x16 tiles per chunk; split-K over 32 blocks,
// f32 atomicAdd reduction.
// ---------------------------------------------------------------------------
#define GRAM_ROWSTRIDE 264            // halfs: 256 data + 8 pad
#define GRAM_CHUNK 256                // columns per TDM tile
#define GRAM_NCH 8                    // chunks per block (2048 cols)

#if HAVE_TDM
static __device__ __forceinline__ void tdm_load_tile(const _Float16* gptr,
                                                     unsigned lds_byte_off) {
  unsigned long long ga = (unsigned long long)(uintptr_t)gptr;
  u32x4 g0;
  g0[0] = 1u;                                   // count=1, user mode
  g0[1] = lds_byte_off;                         // lds_addr
  g0[2] = (unsigned)ga;                         // global_addr[31:0]
  g0[3] = (unsigned)((ga >> 32) & 0x01FFFFFFu)  // global_addr[56:32]
          | 0x80000000u;                        // type=2 ("image")
  i32x8 g1;
  g1[0] = 0x07910000;        // data_size=2B | pad_en | interval=512B | pad=16B
  g1[1] = 0;                 // tensor_dim0[15:0]=0 (dim0 = 0x40000000)
  g1[2] = 0x00304000;        // tensor_dim0[31:16]=0x4000 | tensor_dim1=48
  g1[3] = (GRAM_CHUNK << 16);// tile_dim0 = 256 elements
  g1[4] = 48;                // tile_dim1 = 48 rows
  g1[5] = HW;                // tensor_dim0_stride = 65536
  g1[6] = 0;
  g1[7] = 0;
  i32x4 z4 = {0, 0, 0, 0};
  i32x8 z8 = {0, 0, 0, 0, 0, 0, 0, 0};
  // clang-23 / therock-10.0 arity: (g0, g1, g2, g3, g4, cpol)
  __builtin_amdgcn_tensor_load_to_lds(g0, g1, z4, z4, z8, 0);
}
#endif

__global__ __launch_bounds__(256) void gram_kernel(
    const _Float16* __restrict__ qkv, float* __restrict__ S) {
  __shared__ alignas(16) _Float16 smem[2][2][48 * GRAM_ROWSTRIDE];
  const int bh = blockIdx.z, b = bh >> 2, h = bh & 3;
  const int tid = threadIdx.x, lane = tid & 31, wv = tid >> 5;
  const _Float16* qp = qkv + ((size_t)b * C3 + h * HEAD_DIM) * HW;
  const _Float16* kp = qkv + ((size_t)b * C3 + DIMC + h * HEAD_DIM) * HW;
  const size_t colbase = (size_t)blockIdx.x * (GRAM_NCH * GRAM_CHUNK);

  v8f acc[3][3] = {};

#if HAVE_TDM
  if (wv == 0) {  // prologue: chunk 0 into buffer 0
    tdm_load_tile(qp + colbase, (unsigned)(uintptr_t)&smem[0][0][0]);
    tdm_load_tile(kp + colbase, (unsigned)(uintptr_t)&smem[0][1][0]);
  }
#endif
  for (int c = 0; c < GRAM_NCH; ++c) {
    const int cur = c & 1;
#if HAVE_TDM
    if (wv == 0) __builtin_amdgcn_s_wait_tensorcnt(0);
    __syncthreads();                       // publish buf[cur]
    if (wv == 0 && c + 1 < GRAM_NCH) {     // prefetch next chunk via TDM
      size_t n1 = colbase + (size_t)(c + 1) * GRAM_CHUNK;
      tdm_load_tile(qp + n1, (unsigned)(uintptr_t)&smem[(c + 1) & 1][0][0]);
      tdm_load_tile(kp + n1, (unsigned)(uintptr_t)&smem[(c + 1) & 1][1][0]);
    }
#else
    __syncthreads();
    {  // cooperative 128-bit copy fallback
      size_t n0 = colbase + (size_t)c * GRAM_CHUNK;
      for (int i = tid; i < 2 * 48 * (GRAM_CHUNK / 8); i += 256) {
        int mtx = (i >= 48 * (GRAM_CHUNK / 8));
        int j   = i - mtx * 48 * (GRAM_CHUNK / 8);
        int row = j >> 5, c8 = (j & 31) * 8;
        const _Float16* src = (mtx ? kp : qp) + (size_t)row * HW + n0 + c8;
        *(uint4*)&smem[cur][mtx][row * GRAM_ROWSTRIDE + c8] =
            *(const uint4*)src;
      }
    }
    __syncthreads();
#endif
    const _Float16* qt = &smem[cur][0][0];
    const _Float16* kt = &smem[cur][1][0];
    const int klocal = wv * 32;            // this wave's 32-col slab
    v16h af[3], bf[3];
    #pragma unroll
    for (int i = 0; i < 3; ++i) {
      af[i] = load_a_frag(qt + i * 16 * GRAM_ROWSTRIDE + klocal, GRAM_ROWSTRIDE, lane);
      bf[i] = load_b_frag(kt + i * 16 * GRAM_ROWSTRIDE + klocal, GRAM_ROWSTRIDE, lane);
    }
    #pragma unroll
    for (int i = 0; i < 3; ++i)
      #pragma unroll
      for (int j = 0; j < 3; ++j)
        acc[i][j] = WMMA_F16(af[i], bf[j], acc[i][j]);
    __syncthreads();                       // done reading buf[cur]
  }

  const int m_add = (lane < 16) ? 0 : 8;
  const int n     = lane & 15;
  #pragma unroll
  for (int i = 0; i < 3; ++i)
    #pragma unroll
    for (int j = 0; j < 3; ++j) {
      float* sp = S + ((size_t)bh * HEAD_DIM + i * 16) * HEAD_DIM + j * 16 + n;
      #pragma unroll
      for (int r = 0; r < 8; ++r)
        atomicAdd(&sp[(size_t)(r + m_add) * HEAD_DIM], acc[i][j][r]);
    }
}

// ---------------------------------------------------------------------------
// K4: softmax over e of S * temp[h] / (max(|q_d|,eps)*max(|k_e|,eps)).
// ---------------------------------------------------------------------------
__global__ __launch_bounds__(64) void softmax_kernel(
    const float* __restrict__ S, const float* __restrict__ norms2,
    const float* __restrict__ temp, float* __restrict__ A) {
  __shared__ float buf[64];
  __shared__ float mx, sm;
  const int row = blockIdx.x;               // bh*48 + d
  const int bh = row / HEAD_DIM, d = row % HEAD_DIM;
  const int b = bh >> 2, h = bh & 3;
  const int e = threadIdx.x;
  const float t  = temp[h];
  const float qn = fmaxf(sqrtf(norms2[b * 2 * DIMC + h * HEAD_DIM + d]), 1e-12f);
  float s = -1e30f;
  if (e < HEAD_DIM) {
    float kn = fmaxf(sqrtf(norms2[b * 2 * DIMC + DIMC + h * HEAD_DIM + e]), 1e-12f);
    s = S[(size_t)row * HEAD_DIM + e] * t / (qn * kn);
  }
  buf[e] = s;
  __syncthreads();
  if (e == 0) {
    float m = -1e30f;
    for (int i = 0; i < HEAD_DIM; ++i) m = fmaxf(m, buf[i]);
    mx = m;
  }
  __syncthreads();
  float ex = (e < HEAD_DIM) ? expf(s - mx) : 0.f;
  buf[e] = ex;
  __syncthreads();
  if (e == 0) {
    float t2 = 0.f;
    for (int i = 0; i < HEAD_DIM; ++i) t2 += buf[i];
    sm = t2;
  }
  __syncthreads();
  if (e < HEAD_DIM) A[(size_t)row * HEAD_DIM + e] = ex / sm;
}

// ---------------------------------------------------------------------------
// K5: out_av[b, h*48+d, p] = sum_e attn[b,h,d,e] * v[b,h,e,p].
// Per head: A = attn (K=48 zero-padded to 64), B = v tile transposed in LDS.
// ---------------------------------------------------------------------------
__global__ __launch_bounds__(256) void av_kernel(
    const _Float16* __restrict__ qkv, const float* __restrict__ A,
    _Float16* __restrict__ out) {
  __shared__ alignas(16) _Float16 al[HEAD_DIM][64 + 8];
  __shared__ alignas(16) _Float16 vt[64][64 + 8];
  const int tid = threadIdx.x, lane = tid & 31, wv = tid >> 5;
  const int p0 = blockIdx.x * 64;
  const int b  = blockIdx.y;

  for (int h = 0; h < HEADS; ++h) {
    __syncthreads();
    for (int i = tid; i < HEAD_DIM * 64; i += 256) {
      int d = i >> 6, e = i & 63;
      float v = (e < HEAD_DIM)
                    ? A[((size_t)(b * HEADS + h) * HEAD_DIM + d) * HEAD_DIM + e]
                    : 0.f;
      al[d][e] = (_Float16)v;
    }
    const _Float16* vp = qkv + ((size_t)b * C3 + 2 * DIMC + h * HEAD_DIM) * HW + p0;
    for (int i = tid; i < HEAD_DIM * 8; i += 256) {  // 128-bit v loads
      int e = i >> 3, p8 = (i & 7) * 8;
      FragH u;
      u.q = *(const uint4*)(vp + (size_t)e * HW + p8);
      #pragma unroll
      for (int j = 0; j < 8; ++j) vt[p8 + j][e] = u.h[j];
    }
    for (int i = tid; i < 64 * 16; i += 256) {       // zero pad e = 48..63
      int px = i & 63, e = HEAD_DIM + (i >> 6);
      vt[px][e] = (_Float16)0.f;
    }
    __syncthreads();
    for (int t = wv; t < 12; t += 8) {
      int d0 = (t >> 2) * 16, n0 = (t & 3) * 16;
      v8f acc = {};
      #pragma unroll
      for (int k0 = 0; k0 < 64; k0 += 32) {
        v16h a  = load_a_frag(&al[d0][k0], 64 + 8, lane);
        v16h bb = load_b_frag(&vt[n0][k0], 64 + 8, lane);
        acc = WMMA_F16(a, bb, acc);
      }
      const int m_add = (lane < 16) ? 0 : 8;
      const int n     = lane & 15;
      size_t obase = ((size_t)b * DIMC + h * HEAD_DIM + d0) * HW + p0 + n0 + n;
      #pragma unroll
      for (int r = 0; r < 8; ++r)
        out[obase + (size_t)(r + m_add) * HW] = (_Float16)acc[r];
    }
  }
}

// ---------------------------------------------------------------------------
// K6: proj 1x1 conv == GEMM  y[b,o,p] = sum_c Wp[o,c] * out_av[b,c,p].
// ---------------------------------------------------------------------------
__global__ __launch_bounds__(256) void proj_kernel(
    const _Float16* __restrict__ in, const float* __restrict__ w,
    float* __restrict__ y) {
  __shared__ alignas(16) _Float16 wl[32][DIMC + 8];
  __shared__ alignas(16) _Float16 xt[64][32 + 8];
  const int tid  = threadIdx.x;
  const int lane = tid & 31;
  const int wv   = tid >> 5;
  const int wr   = wv >> 2;
  const int wc   = wv & 3;
  const int p0   = blockIdx.x * 64;
  const int oc0  = blockIdx.y * 32;
  const int b    = blockIdx.z;

  for (int i = tid; i < 32 * (DIMC / 4); i += 256) {
    int r = i / (DIMC / 4), c4 = (i % (DIMC / 4)) * 4;
    float4 v = *(const float4*)(w + (size_t)(oc0 + r) * DIMC + c4);
    wl[r][c4 + 0] = (_Float16)v.x;
    wl[r][c4 + 1] = (_Float16)v.y;
    wl[r][c4 + 2] = (_Float16)v.z;
    wl[r][c4 + 3] = (_Float16)v.w;
  }
  const _Float16* xb = in + (size_t)b * DIMC * HW;

  v8f acc = {};
  for (int k0 = 0; k0 < DIMC; k0 += 32) {
    __syncthreads();
    {  // one 128-bit load per thread covers the 32x64 f16 chunk
      int h8 = tid & 7;
      int k  = tid >> 3;   // 0..31
      FragH u;
      u.q = *(const uint4*)(xb + (size_t)(k0 + k) * HW + p0 + h8 * 8);
      #pragma unroll
      for (int j = 0; j < 8; ++j) xt[h8 * 8 + j][k] = u.h[j];
    }
    __syncthreads();
    v16h a  = load_a_frag(&wl[wr * 16][k0], DIMC + 8, lane);
    v16h bb = load_b_frag(&xt[wc * 16][0], 32 + 8, lane);
    acc = WMMA_F16(a, bb, acc);
  }
  const int m_add = (lane < 16) ? 0 : 8;
  const int n     = lane & 15;
  size_t obase = ((size_t)b * DIMC + oc0 + wr * 16) * HW + p0 + wc * 16 + n;
  #pragma unroll
  for (int r = 0; r < 8; ++r)
    y[obase + (size_t)(r + m_add) * HW] = acc[r];
}

// ---------------------------------------------------------------------------
// Workspace layout (bytes):
//   [0,   302MB) : tmp1  f16 qkv pre-dwconv; REUSED later as out_av (100MB)
//   [302, 604MB) : qkvb  f16 qkv post-dwconv
//   [604MB, ..)  : norms2 f32 [B][2*192], S f32 [16][48][48], Afin f32 same
// ---------------------------------------------------------------------------
extern "C" void kernel_launch(void* const* d_in, const int* in_sizes, int n_in,
                              void* d_out, int out_size, void* d_ws,
                              size_t ws_size, hipStream_t stream) {
  const float* x      = (const float*)d_in[0];
  const float* qkv_w  = (const float*)d_in[1];
  const float* dw_w   = (const float*)d_in[2];
  const float* proj_w = (const float*)d_in[3];
  const float* temper = (const float*)d_in[4];
  float* y = (float*)d_out;

  const size_t OFF1 = (size_t)C3 * BATCH * HW * sizeof(_Float16);
  char* ws = (char*)d_ws;
  _Float16* tmp1   = (_Float16*)ws;
  _Float16* qkvb   = (_Float16*)(ws + OFF1);
  float*    norms2 = (float*)(ws + 2 * OFF1);
  float*    S      = norms2 + BATCH * 2 * DIMC;
  float*    Afin   = S + 16 * HEAD_DIM * HEAD_DIM;
  _Float16* outav  = tmp1;  // tmp1 dead after dwconv; reuse

  const int nzero = BATCH * 2 * DIMC + 16 * HEAD_DIM * HEAD_DIM;
  zero_kernel<<<(nzero + 255) / 256, 256, 0, stream>>>(norms2, nzero);

  qkv1x1_kernel<<<dim3(HW / 64, C3 / 32, BATCH), 256, 0, stream>>>(x, qkv_w, tmp1);
  dwconv_kernel<<<dim3((HGT / 32) * (WID / 32), C3, BATCH), 256, 0, stream>>>(
      tmp1, dw_w, qkvb, norms2);
  gram_kernel<<<dim3(HW / (GRAM_NCH * GRAM_CHUNK), 1, BATCH * HEADS), 256, 0,
                stream>>>(qkvb, S);
  softmax_kernel<<<dim3(BATCH * HEADS * HEAD_DIM), 64, 0, stream>>>(
      S, norms2, temper, Afin);
  av_kernel<<<dim3(HW / 64, BATCH), 256, 0, stream>>>(qkvb, Afin, outav);
  proj_kernel<<<dim3(HW / 64, DIMC / 32, BATCH), 256, 0, stream>>>(outav, proj_w, y);
}
